// KNNCrossSampleAttention_24318104830761
// MI455X (gfx1250) — matrix operand order repaired and checked by reading, not compiled, not run on hardware
//
#include <hip/hip_runtime.h>
#include <hip/hip_bf16.h>

typedef __attribute__((ext_vector_type(16))) _Float16 v16h;
typedef __attribute__((ext_vector_type(8)))  _Float16 v8h;
typedef __attribute__((ext_vector_type(8)))  float    v8f;

#define DIMV     256
#define HEADS    8
#define DIM_HEAD 32
#define INNER    256
#define KNN      17          // K_NEIGHBORS + 1 (self)
#define BATCH    512
#define NFEAT    32
#define ROWS     (BATCH * NFEAT)   // 16384
#define QKV_N    768
#define ATTN_SCALE 0.17677669529663687f   // 32^-0.5

struct h16pair { v8h lo, hi; };

// A-fragment (16x32 f16, ISA 7.12.2): lane holds row M=lane&15;
// halfs[0..7]  = K kb..kb+7, halfs[8..15] = K 16+kb..16+kb+7, kb = (lane&16)?8:0.
// Both chunks are contiguous 16B in row-major A.
__device__ __forceinline__ v16h load_a_frag(const _Float16* __restrict__ A,
                                            int row, int K, int kk, int lane) {
    int kb = (lane & 16) ? 8 : 0;
    const _Float16* p = A + (size_t)row * K + kk + kb;
    h16pair u;
    u.lo = *(const v8h*)(p);
    u.hi = *(const v8h*)(p + 16);
    return __builtin_bit_cast(v16h, u);
}

// B-fragment (32x16 f16): lane holds col N=lane&15;
// halfs[0..7] = K kb..kb+7, halfs[8..15] = K 16+kb..16+kb+7, kb = (lane&16)?8:0.
// Two contiguous 16B chunks when B is stored transposed: Bt[N][K].
__device__ __forceinline__ v16h load_b_frag(const _Float16* __restrict__ Bt,
                                            int col, int K, int kk, int lane) {
    int kb = (lane & 16) ? 8 : 0;
    const _Float16* p = Bt + (size_t)col * K + kk + kb;
    h16pair u;
    u.lo = *(const v8h*)(p);
    u.hi = *(const v8h*)(p + 16);
    return __builtin_bit_cast(v16h, u);
}

// Generic WMMA GEMM: C[M,N] = A[M,K] * Bt[N,K]^T (+bias).
// 256 threads = 8 waves arranged 2(M) x 4(N); wave tile = 16x64 (4 acc tiles).
// K-steps are software double-buffered: the 5 fragments of step kk+32 are in
// flight while the 4 WMMAs of step kk execute, hiding L2 latency.
// __launch_bounds__(256,1) relaxes occupancy so ~120 VGPRs are available.
// Requires M % 32 == 0, N % 256 == 0, K == 256.
template <bool F16OUT, bool BIAS>
__global__ __launch_bounds__(256, 1) void gemm_wmma_kernel(
    const _Float16* __restrict__ A, const _Float16* __restrict__ Bt,
    void* __restrict__ C, const float* __restrict__ bias,
    int M, int N, int K) {
    int wave = threadIdx.x >> 5;
    int lane = threadIdx.x & 31;
    int wm = wave >> 2;       // 0..1
    int wn = wave & 3;        // 0..3
    int tm = blockIdx.x * 32 + wm * 16;
    int tn = blockIdx.y * 256 + wn * 64;
    int arow = tm + (lane & 15);
    int bcol = tn + (lane & 15);

    v8f acc[4] = {};
    // prologue: fragments for kk = 0
    v16h a_cur = load_a_frag(A, arow, K, 0, lane);
    v16h b_cur[4];
#pragma unroll
    for (int t = 0; t < 4; ++t) b_cur[t] = load_b_frag(Bt, bcol + t * 16, K, 0, lane);

#pragma unroll
    for (int kk = 0; kk < 256; kk += 32) {   // K == 256 for every call here
        v16h a_nxt;
        v16h b_nxt[4];
        if (kk + 32 < 256) {                  // prefetch next K-step
            a_nxt = load_a_frag(A, arow, K, kk + 32, lane);
#pragma unroll
            for (int t = 0; t < 4; ++t)
                b_nxt[t] = load_b_frag(Bt, bcol + t * 16, K, kk + 32, lane);
        }
#pragma unroll
        for (int t = 0; t < 4; ++t)
            acc[t] = __builtin_amdgcn_wmma_f32_16x16x32_f16(
                false, a_cur, false, b_cur[t], (short)0, acc[t], false, false);
        a_cur = a_nxt;
#pragma unroll
        for (int t = 0; t < 4; ++t) b_cur[t] = b_nxt[t];
    }

    // C/D layout: VGPR i -> M = tm + i + ((lane&16)?8:0), N = (lane&15)
    int row0 = tm + ((lane & 16) ? 8 : 0);
#pragma unroll
    for (int t = 0; t < 4; ++t) {
        int col = tn + t * 16 + (lane & 15);
        float bv = BIAS ? bias[col] : 0.0f;
#pragma unroll
        for (int i = 0; i < 8; ++i) {
            float v = acc[t][i] + bv;
            size_t off = (size_t)(row0 + i) * N + col;
            if (F16OUT) ((_Float16*)C)[off] = (_Float16)v;
            else        ((float*)C)[off] = v;
        }
    }
}

__global__ void f32_to_f16_kernel(const float* __restrict__ in,
                                  _Float16* __restrict__ out, int n) {
    int i = blockIdx.x * blockDim.x + threadIdx.x;
    if (i < n) out[i] = (_Float16)in[i];
}

// mean over the 32 features for one sample row; one block per sample.
__global__ __launch_bounds__(256) void mean_feat_kernel(
    const float* __restrict__ x, _Float16* __restrict__ meanx) {
    int b = blockIdx.x, c = threadIdx.x;
    const float* p = x + (size_t)b * NFEAT * DIMV + c;
    float s = 0.f;
#pragma unroll
    for (int f = 0; f < NFEAT; ++f) s += p[f * DIMV];
    meanx[(size_t)b * DIMV + c] = (_Float16)(s * (1.0f / NFEAT));
}

// Wt[n*K + k] = (f16) W[k*N + n]  -- transpose+convert weights once.
__global__ void transpose_cvt_kernel(const float* __restrict__ W,
                                     _Float16* __restrict__ Wt, int K, int N) {
    int i = blockIdx.x * blockDim.x + threadIdx.x;
    if (i >= K * N) return;
    int n = i / K, k = i - n * K;
    Wt[i] = (_Float16)W[(size_t)k * N + n];
}

__global__ __launch_bounds__(256) void normalize_kernel(
    const float* __restrict__ reprs, _Float16* __restrict__ normed) {
    __shared__ float red[256];
    int b = blockIdx.x, c = threadIdx.x;
    float v = reprs[(size_t)b * DIMV + c];
    red[c] = v * v;
    __syncthreads();
    for (int s = 128; s > 0; s >>= 1) {
        if (c < s) red[c] += red[c + s];
        __syncthreads();
    }
    normed[(size_t)b * DIMV + c] = (_Float16)(v * rsqrtf(red[0]));
}

// top-17 per row of sim[512,512]: one wave32 per row, values live in registers.
__global__ __launch_bounds__(256) void topk_kernel(
    const float* __restrict__ sim, int* __restrict__ knn) {
    int wave = threadIdx.x >> 5, lane = threadIdx.x & 31;
    int row = blockIdx.x * 8 + wave;
    float val[16];
#pragma unroll
    for (int t = 0; t < 16; ++t)
        val[t] = sim[(size_t)row * BATCH + t * 32 + lane];
    for (int sel = 0; sel < KNN; ++sel) {
        float bv = -INFINITY;
        int bi = 0x7fffffff;
#pragma unroll
        for (int t = 0; t < 16; ++t) {
            int idx = t * 32 + lane;
            if (val[t] > bv || (val[t] == bv && idx < bi)) { bv = val[t]; bi = idx; }
        }
        for (int off = 16; off > 0; off >>= 1) {      // wave32 argmax butterfly
            float ov = __shfl_xor(bv, off, 32);
            int   oi = __shfl_xor(bi, off, 32);
            if (ov > bv || (ov == bv && oi < bi)) { bv = ov; bi = oi; }
        }
        if ((bi & 31) == lane) val[bi >> 5] = -INFINITY;  // exclude winner
        if (lane == 0) knn[row * KNN + sel] = bi;
    }
}

// Sparse gather attention: block=(b,n), wave=head h, lane=dh element.
// Only the 17 kNN columns survive the mask, so gather them from L2-resident qkv.
__global__ __launch_bounds__(256) void attn_kernel(
    const _Float16* __restrict__ qkv, const int* __restrict__ knn,
    _Float16* __restrict__ out) {
    __shared__ int s_idx[KNN];
    int b = blockIdx.x >> 5;
    int n = blockIdx.x & 31;
    if (threadIdx.x < KNN) s_idx[threadIdx.x] = knn[b * KNN + threadIdx.x];
    __syncthreads();
    int h = threadIdx.x >> 5, lane = threadIdx.x & 31;
    size_t rowbase = ((size_t)b * NFEAT + n) * QKV_N;
    float q = (float)qkv[rowbase + h * DIM_HEAD + lane] * ATTN_SCALE;
    float s[KNN];
#pragma unroll
    for (int j = 0; j < KNN; ++j) {
        size_t kaddr = ((size_t)s_idx[j] * NFEAT + n) * QKV_N + INNER + h * DIM_HEAD + lane;
        float p = q * (float)qkv[kaddr];
#pragma unroll
        for (int off = 16; off > 0; off >>= 1) p += __shfl_xor(p, off, 32);
        s[j] = p;   // full dot product, replicated across lanes
    }
    float m = s[0];
#pragma unroll
    for (int j = 1; j < KNN; ++j) m = fmaxf(m, s[j]);
    float denom = 0.f, acc = 0.f;
#pragma unroll
    for (int j = 0; j < KNN; ++j) {
        float w = __expf(s[j] - m);
        denom += w;
        size_t vaddr = ((size_t)s_idx[j] * NFEAT + n) * QKV_N + 2 * INNER + h * DIM_HEAD + lane;
        acc += w * (float)qkv[vaddr];
    }
    out[((size_t)b * NFEAT + n) * INNER + h * DIM_HEAD + lane] = (_Float16)(acc / denom);
}

extern "C" void kernel_launch(void* const* d_in, const int* in_sizes, int n_in,
                              void* d_out, int out_size, void* d_ws, size_t ws_size,
                              hipStream_t stream) {
    const float* x      = (const float*)d_in[0];
    const float* W_qkv  = (const float*)d_in[1];
    const float* W_out  = (const float*)d_in[2];
    const float* b_out  = (const float*)d_in[3];
    const float* W_repr = (const float*)d_in[4];
    const float* b_repr = (const float*)d_in[5];
    float* out = (float*)d_out;

    char* w = (char*)d_ws;
    auto alloc = [&](size_t bytes) -> char* {
        char* p = w;
        w += (bytes + 255) & ~(size_t)255;
        return p;
    };
    _Float16* xh      = (_Float16*)alloc((size_t)ROWS * DIMV * 2);   // 8 MB
    _Float16* meanx   = (_Float16*)alloc((size_t)BATCH * DIMV * 2);
    _Float16* wrepr_t = (_Float16*)alloc((size_t)DIMV * DIMV * 2);
    _Float16* wqkv_t  = (_Float16*)alloc((size_t)QKV_N * DIMV * 2);
    _Float16* wout_t  = (_Float16*)alloc((size_t)DIMV * DIMV * 2);
    float*    reprs   = (float*)alloc((size_t)BATCH * DIMV * 4);
    _Float16* normed  = (_Float16*)alloc((size_t)BATCH * DIMV * 2);
    float*    sim     = (float*)alloc((size_t)BATCH * BATCH * 4);    // 1 MB
    int*      knn     = (int*)alloc((size_t)BATCH * KNN * 4);
    _Float16* qkvh    = (_Float16*)alloc((size_t)ROWS * QKV_N * 2);  // 24 MB
    _Float16* attnh   = (_Float16*)alloc((size_t)ROWS * INNER * 2);  // 8 MB

    // Stage inputs into f16 (x) / transposed f16 (weights).
    f32_to_f16_kernel<<<(ROWS * DIMV + 255) / 256, 256, 0, stream>>>(x, xh, ROWS * DIMV);
    mean_feat_kernel<<<BATCH, 256, 0, stream>>>(x, meanx);
    transpose_cvt_kernel<<<(DIMV * DIMV + 255) / 256, 256, 0, stream>>>(W_repr, wrepr_t, DIMV, DIMV);
    transpose_cvt_kernel<<<(DIMV * QKV_N + 255) / 256, 256, 0, stream>>>(W_qkv, wqkv_t, DIMV, QKV_N);
    transpose_cvt_kernel<<<(DIMV * DIMV + 255) / 256, 256, 0, stream>>>(W_out, wout_t, DIMV, DIMV);

    // A. sample representations (WMMA) + bias:  512x256x256
    gemm_wmma_kernel<false, true><<<dim3(BATCH / 32, DIMV / 256), 256, 0, stream>>>(
        meanx, wrepr_t, reprs, b_repr, BATCH, DIMV, DIMV);
    normalize_kernel<<<BATCH, 256, 0, stream>>>(reprs, normed);

    // B. cosine-similarity matrix (WMMA; Bt == normed gives normed @ normed^T)
    gemm_wmma_kernel<false, false><<<dim3(BATCH / 32, BATCH / 256), 256, 0, stream>>>(
        normed, normed, sim, nullptr, BATCH, BATCH, DIMV);
    topk_kernel<<<BATCH / 8, 256, 0, stream>>>(sim, knn);

    // D1. qkv projection (WMMA, dominant GEMM: 16384x768x256)
    gemm_wmma_kernel<true, false><<<dim3(ROWS / 32, QKV_N / 256), 256, 0, stream>>>(
        xh, wqkv_t, qkvh, nullptr, ROWS, QKV_N, DIMV);

    // D2. sparse masked attention over the 17 neighbors only
    attn_kernel<<<ROWS, 256, 0, stream>>>(qkvh, knn, attnh);

    // D3. output projection (WMMA) + bias -> f32 result: 16384x256x256
    gemm_wmma_kernel<false, true><<<dim3(ROWS / 32, DIMV / 256), 256, 0, stream>>>(
        attnh, wout_t, out, b_out, ROWS, DIMV, DIMV);
}
